// MultiScaleAttention_53730040873118
// MI455X (gfx1250) — matrix operand order repaired and checked by reading, not compiled, not run on hardware
//
#include <hip/hip_runtime.h>

// ---------------------------------------------------------------------------
// MultiScaleAttention (MViT block) for gfx1250 — WMMA f16 path, f32 accum.
// B=2, DIM=768, NH=12, HD=64, N=6273 (1+8*28*28), KN=393 (1+8*7*7)
// ---------------------------------------------------------------------------

#define Bc      2
#define DIM     768
#define NHd     12
#define HDc     64
#define Nt      6273        // q tokens incl cls
#define QTt     8
#define QHh     28
#define QWw     28
#define KNt     393         // k tokens incl cls
#define KNP     416         // padded to 13*32 (WMMA K multiple)
#define MQP     6288        // per-(b,h) q tokens padded to 16 (393 tiles)
#define MROWS   (Bc*Nt)     // 12546
#define MPAD    12576       // padded to 32 (393 tiles of 32)
#define SCALEc  0.125f

typedef __attribute__((ext_vector_type(16))) _Float16 v16h;
typedef __attribute__((ext_vector_type(8)))  _Float16 v8h;
typedef __attribute__((ext_vector_type(8)))  float    v8f;

__device__ __forceinline__ v8f wmma_f16(v16h a, v16h b, v8f c) {
  // D = A(16x32 f16) * B(32x16 f16) + C(16x16 f32)
  return __builtin_amdgcn_wmma_f32_16x16x32_f16(
      /*neg_a=*/false, a, /*neg_b=*/false, b,
      /*c_mod=*/(short)0, c, /*reuse_a=*/false, /*reuse_b=*/false);
}

// -------------------------------- converts ---------------------------------
__global__ void f32_to_f16_pad(const float* __restrict__ in,
                               _Float16* __restrict__ out,
                               long nin, long nout) {
  long i = (long)blockIdx.x * blockDim.x + threadIdx.x;
  if (i < nout) out[i] = (i < nin) ? (_Float16)in[i] : (_Float16)0.0f;
}

__global__ void zero_f16(_Float16* __restrict__ p, long n) {
  long i = (long)blockIdx.x * blockDim.x + threadIdx.x;
  if (i < n) p[i] = (_Float16)0.0f;
}

// --------------------------- generic WMMA GEMM -----------------------------
// D(Mrows x Nld, f32, row-major) = A(MPADrows x K, f16 row-major)
//                                  * Wt(N x K, f16 row-major)^T + bias(N)
// One wave per 32(M) x 64(N) tile: two A fragments share four B fragments,
// so each B load feeds 2 WMMAs (8 WMMAs per 32-wide k-step).
// A must be allocated/zeroed out to the padded M row count.
__global__ __launch_bounds__(32)
void wmma_gemm_bias(const _Float16* __restrict__ A,
                    const _Float16* __restrict__ Wt,
                    const float* __restrict__ bias,
                    float* __restrict__ D,
                    int Mrows, int K, int Nld) {
  const int lane = threadIdx.x & 31;
  const int m    = lane & 15;
  const int half = lane >> 4;
  const int row0 = blockIdx.x * 32;
  const int col0 = blockIdx.y * 64;

  v8f acc[2][4] = {};
  const _Float16* arow0 = A + (size_t)(row0 + m) * K;
  const _Float16* arow1 = A + (size_t)(row0 + 16 + m) * K;
  const _Float16* wrow  = Wt + (size_t)(col0 + m) * K + half * 16;

  for (int k0 = 0; k0 < K; k0 += 32) {
    union { v16h v; v8h h[2]; } ua0, ua1;
    ua0.h[0] = *(const v8h*)(arow0 + k0 + half * 16);
    ua0.h[1] = *(const v8h*)(arow0 + k0 + half * 16 + 8);
    ua1.h[0] = *(const v8h*)(arow1 + k0 + half * 16);
    ua1.h[1] = *(const v8h*)(arow1 + k0 + half * 16 + 8);
    v16h b0 = *(const v16h*)(wrow + k0);
    v16h b1 = *(const v16h*)(wrow + (size_t)16 * K + k0);
    v16h b2 = *(const v16h*)(wrow + (size_t)32 * K + k0);
    v16h b3 = *(const v16h*)(wrow + (size_t)48 * K + k0);
    acc[0][0] = wmma_f16(ua0.v, b0, acc[0][0]);
    acc[1][0] = wmma_f16(ua1.v, b0, acc[1][0]);
    acc[0][1] = wmma_f16(ua0.v, b1, acc[0][1]);
    acc[1][1] = wmma_f16(ua1.v, b1, acc[1][1]);
    acc[0][2] = wmma_f16(ua0.v, b2, acc[0][2]);
    acc[1][2] = wmma_f16(ua1.v, b2, acc[1][2]);
    acc[0][3] = wmma_f16(ua0.v, b3, acc[0][3]);
    acc[1][3] = wmma_f16(ua1.v, b3, acc[1][3]);
  }

  const int n = m;
  const float bi0 = bias[col0 + n];
  const float bi1 = bias[col0 + 16 + n];
  const float bi2 = bias[col0 + 32 + n];
  const float bi3 = bias[col0 + 48 + n];
#pragma unroll
  for (int sub = 0; sub < 2; ++sub) {
#pragma unroll
    for (int v = 0; v < 8; ++v) {
      int r = row0 + sub * 16 + half * 8 + v;
      if (r < Mrows) {
        float* drow = D + (size_t)r * Nld + col0;
        drow[n]      = acc[sub][0][v] + bi0;
        drow[16 + n] = acc[sub][1][v] + bi1;
        drow[32 + n] = acc[sub][2][v] + bi2;
        drow[48 + n] = acc[sub][3][v] + bi3;
      }
    }
  }
}

// ----------------------- pool (depthwise conv) + LN ------------------------
// qkv32: (B*Nt, 2304) f32. Output f16: either (24, tokpad, 64) [normal]
// or (24, 64, tokpad) [transposed, for V^T]. Pads written as zeros.
__global__ __launch_bounds__(64)
void pool_ln_kernel(const float* __restrict__ qkv,
                    const float* __restrict__ w,   // (3,3,3,1,64) flat
                    const float* __restrict__ g,
                    const float* __restrict__ bta,
                    _Float16* __restrict__ out,
                    int s_idx, int sp, int OT, int OH, int OW,
                    int tokpad, int transposed) {
  const int d   = threadIdx.x;     // 0..63
  const int tok = blockIdx.x;      // 0..tokpad-1
  const int bn  = blockIdx.y;      // 0..23
  const int b   = bn / NHd;
  const int nh  = bn % NHd;
  const int ntok = 1 + OT * OH * OW;
  const int colbase = s_idx * DIM + nh * HDc + d;

  float val = 0.0f;
  if (tok < ntok) {
    if (tok == 0) {
      val = qkv[(size_t)(b * Nt) * (3 * DIM) + colbase];
    } else {
      int bi = tok - 1;
      int to = bi / (OH * OW);
      int r  = bi % (OH * OW);
      int ho = r / OW, wo = r % OW;
#pragma unroll
      for (int dt = 0; dt < 3; ++dt) {
        int ti = to + dt - 1;
        if (ti < 0 || ti >= QTt) continue;
#pragma unroll
        for (int dh = 0; dh < 3; ++dh) {
          int hi = ho * sp + dh - 1;
          if (hi < 0 || hi >= QHh) continue;
#pragma unroll
          for (int dw = 0; dw < 3; ++dw) {
            int wi = wo * sp + dw - 1;
            if (wi < 0 || wi >= QWw) continue;
            int itok = 1 + (ti * QHh + hi) * QWw + wi;
            val += qkv[(size_t)(b * Nt + itok) * (3 * DIM) + colbase] *
                   w[((dt * 3 + dh) * 3 + dw) * HDc + d];
          }
        }
      }
    }
  }

  __shared__ float s1[64], s2[64];
  s1[d] = val; s2[d] = val * val;
  __syncthreads();
  for (int off = 32; off > 0; off >>= 1) {
    if (d < off) { s1[d] += s1[d + off]; s2[d] += s2[d + off]; }
    __syncthreads();
  }
  float mean = s1[0] * (1.0f / 64.0f);
  float var  = s2[0] * (1.0f / 64.0f) - mean * mean;
  float ln = (tok < ntok)
               ? ((val - mean) * rsqrtf(var + 1e-6f) * g[d] + bta[d])
               : 0.0f;

  if (!transposed)
    out[((size_t)bn * tokpad + tok) * HDc + d] = (_Float16)ln;
  else
    out[((size_t)bn * HDc + d) * tokpad + tok] = (_Float16)ln;
}

// ------------------------------ rel-pos bias -------------------------------
// relt (24,6272,8), relh (24,6272,7), relw (24,6272,7) f32.
__global__ __launch_bounds__(32)
void relpos_kernel(const _Float16* __restrict__ qh,   // (24, MQP, 64) unscaled
                   const float* __restrict__ rph,     // (55,64)
                   const float* __restrict__ rpw,     // (55,64)
                   const float* __restrict__ rpt,     // (15,64)
                   float* __restrict__ relt,
                   float* __restrict__ relh,
                   float* __restrict__ relw) {
  const int lane = threadIdx.x;
  const int qi   = blockIdx.x;   // 0..6271
  const int bn   = blockIdx.y;   // 0..23
  const int t = qi / (QHh * QWw);
  const int r = qi % (QHh * QWw);
  const int h = r / QWw, w = r % QWw;

  __shared__ float qv[64];
  const _Float16* qp = qh + ((size_t)bn * MQP + 1 + qi) * HDc;
  qv[lane]      = (float)qp[lane];
  qv[lane + 32] = (float)qp[lane + 32];
  __syncthreads();

  if (lane < 8) {                       // rel_t, kt = lane
    const float* rp = rpt + (size_t)(t - lane + (QTt - 1)) * HDc;
    float s = 0.0f;
    for (int d = 0; d < 64; ++d) s += qv[d] * rp[d];
    relt[((size_t)bn * (QTt * QHh * QWw) + qi) * 8 + lane] = s;
  } else if (lane < 15) {               // rel_h, k = lane-8
    int k = lane - 8;
    const float* rp = rph + (size_t)(h - 4 * k + 24) * HDc;
    float s = 0.0f;
    for (int d = 0; d < 64; ++d) s += qv[d] * rp[d];
    relh[((size_t)bn * (QTt * QHh * QWw) + qi) * 7 + k] = s;
  } else if (lane >= 16 && lane < 23) { // rel_w, k = lane-16
    int k = lane - 16;
    const float* rp = rpw + (size_t)(w - 4 * k + 24) * HDc;
    float s = 0.0f;
    for (int d = 0; d < 64; ++d) s += qv[d] * rp[d];
    relw[((size_t)bn * (QTt * QHh * QWw) + qi) * 7 + k] = s;
  }
}

// --------------------------- fused attention -------------------------------
// One wave = 16 q-rows of one (b,head). Scores kept in LDS (never spilled
// to HBM): QK WMMA -> scale+bias+mask -> softmax -> f16 probs -> AV WMMA.
__global__ __launch_bounds__(32)
void attn_fused(const _Float16* __restrict__ qh,   // (24, MQP, 64)
                const _Float16* __restrict__ kh,   // (24, KNP, 64)
                const _Float16* __restrict__ vth,  // (24, 64, KNP)
                const float* __restrict__ relt,
                const float* __restrict__ relh,
                const float* __restrict__ relw,
                _Float16* __restrict__ ctx) {      // (MPAD, 768)
  __shared__ __align__(16) float    sc[16][KNP];
  __shared__ __align__(16) _Float16 pr[16][KNP];

  const int lane = threadIdx.x & 31;
  const int m    = lane & 15;
  const int half = lane >> 4;
  const int n    = m;
  const int qt   = blockIdx.x;           // 0..392
  const int bn   = blockIdx.y;           // 0..23
  const int b    = bn / NHd;
  const int nh   = bn % NHd;
  const int row0 = qt * 16;

  // --- A fragments of the q tile (K = 64 -> two k-steps of 32) ---
  const _Float16* qbase = qh + ((size_t)bn * MQP + row0 + m) * HDc;
  union { v16h v; v8h h[2]; } a0, a1;
  a0.h[0] = *(const v8h*)(qbase + half * 16);
  a0.h[1] = *(const v8h*)(qbase + half * 16 + 8);
  a1.h[0] = *(const v8h*)(qbase + 32 + half * 16);
  a1.h[1] = *(const v8h*)(qbase + 32 + half * 16 + 8);

  const float* rtq = relt + (size_t)bn * (QTt * QHh * QWw) * 8;
  const float* rhq = relh + (size_t)bn * (QTt * QHh * QWw) * 7;
  const float* rwq = relw + (size_t)bn * (QTt * QHh * QWw) * 7;

  // --- QK: 26 n-tiles of 16 k-tokens ---
  for (int nt = 0; nt < 26; ++nt) {
    const _Float16* kb =
        kh + ((size_t)bn * KNP + nt * 16 + n) * HDc + half * 16;
    v16h b0 = *(const v16h*)(kb);
    v16h b1 = *(const v16h*)(kb + 32);
    v8f acc = {};
    acc = wmma_f16(a0.v, b0, acc);
    acc = wmma_f16(a1.v, b1, acc);

    const int kcol = nt * 16 + n;
    int kt = 0, kh2 = 0, kw2 = 0;
    const bool kvalid = (kcol >= 1 && kcol < KNt);
    if (kvalid) {
      int ki = kcol - 1;
      kt  = ki / 49;
      int kr = ki % 49;
      kh2 = kr / 7;
      kw2 = kr % 7;
    }
#pragma unroll
    for (int v = 0; v < 8; ++v) {
      int qrow = row0 + half * 8 + v;
      float s;
      if (kcol >= KNt) {
        s = -1e30f;                         // mask padded k columns
      } else {
        s = acc[v] * SCALEc;
        if (kvalid && qrow >= 1 && qrow < Nt) {
          int qi = qrow - 1;
          s += rtq[(size_t)qi * 8 + kt] +
               rhq[(size_t)qi * 7 + kh2] +
               rwq[(size_t)qi * 7 + kw2];
        }
      }
      sc[half * 8 + v][kcol] = s;
    }
  }
  __syncthreads();

  // --- softmax: lane pair (l, l+16) handles row l&15, half the columns ---
  {
    const int row  = m;
    const int cbeg = half * (KNP / 2);
    float mx = -3.0e38f;
    for (int c = 0; c < KNP / 2; ++c) mx = fmaxf(mx, sc[row][cbeg + c]);
    mx = fmaxf(mx, __shfl_xor(mx, 16, 32));
    float ssum = 0.0f;
    for (int c = 0; c < KNP / 2; ++c) {
      float e = __expf(sc[row][cbeg + c] - mx);
      sc[row][cbeg + c] = e;
      ssum += e;
    }
    ssum += __shfl_xor(ssum, 16, 32);
    float inv = 1.0f / ssum;
    for (int c = 0; c < KNP / 2; ++c)
      pr[row][cbeg + c] = (_Float16)(sc[row][cbeg + c] * inv);
  }
  __syncthreads();

  // --- AV: out(16x64) = P(16x416) * V(416x64), V supplied as V^T (N x K) ---
  v8f o0 = {}, o1 = {}, o2 = {}, o3 = {};
  for (int s = 0; s < KNP / 32; ++s) {
    union { v16h v; v8h h[2]; } pa;
    const _Float16* pp = &pr[m][s * 32 + half * 16];
    pa.h[0] = *(const v8h*)(pp);
    pa.h[1] = *(const v8h*)(pp + 8);
    const _Float16* vb =
        vth + ((size_t)bn * HDc + n) * KNP + s * 32 + half * 16;
    v16h b0 = *(const v16h*)(vb);
    v16h b1 = *(const v16h*)(vb + (size_t)16 * KNP);
    v16h b2 = *(const v16h*)(vb + (size_t)32 * KNP);
    v16h b3 = *(const v16h*)(vb + (size_t)48 * KNP);
    o0 = wmma_f16(pa.v, b0, o0);
    o1 = wmma_f16(pa.v, b1, o1);
    o2 = wmma_f16(pa.v, b2, o2);
    o3 = wmma_f16(pa.v, b3, o3);
  }

#pragma unroll
  for (int v = 0; v < 8; ++v) {
    int qrow = row0 + half * 8 + v;
    if (qrow < Nt) {
      _Float16* crow = ctx + (size_t)(b * Nt + qrow) * DIM + nh * HDc;
      crow[n]      = (_Float16)o0[v];
      crow[16 + n] = (_Float16)o1[v];
      crow[32 + n] = (_Float16)o2[v];
      crow[48 + n] = (_Float16)o3[v];
    }
  }
}

// ------------------------------- launcher ----------------------------------
extern "C" void kernel_launch(void* const* d_in, const int* in_sizes, int n_in,
                              void* d_out, int out_size, void* d_ws,
                              size_t ws_size, hipStream_t stream) {
  const float* x      = (const float*)d_in[0];
  const float* qkv_w  = (const float*)d_in[1];
  const float* qkv_b  = (const float*)d_in[2];
  const float* proj_w = (const float*)d_in[3];
  const float* proj_b = (const float*)d_in[4];
  const float* pq_w   = (const float*)d_in[5];
  const float* pk_w   = (const float*)d_in[6];
  const float* pv_w   = (const float*)d_in[7];
  const float* nq_g   = (const float*)d_in[8];
  const float* nq_b   = (const float*)d_in[9];
  const float* nk_g   = (const float*)d_in[10];
  const float* nk_b   = (const float*)d_in[11];
  const float* nv_g   = (const float*)d_in[12];
  const float* nv_b   = (const float*)d_in[13];
  const float* rph    = (const float*)d_in[14];
  const float* rpw    = (const float*)d_in[15];
  const float* rpt    = (const float*)d_in[16];
  float* out = (float*)d_out;

  // workspace carve-up (256-B aligned)
  char* base = (char*)d_ws;
  size_t off = 0;
  auto carve = [&](size_t bytes) -> char* {
    char* p = base + off;
    off = (off + bytes + 255) & ~(size_t)255;
    return p;
  };
  _Float16* xh     = (_Float16*)carve((size_t)MPAD * DIM * 2);
  _Float16* qkvwh  = (_Float16*)carve((size_t)3 * DIM * DIM * 2);
  _Float16* projwh = (_Float16*)carve((size_t)DIM * DIM * 2);
  float*    qkv32  = (float*)   carve((size_t)MROWS * 3 * DIM * 4);
  _Float16* qhB    = (_Float16*)carve((size_t)24 * MQP * HDc * 2);
  _Float16* khB    = (_Float16*)carve((size_t)24 * KNP * HDc * 2);
  _Float16* vthB   = (_Float16*)carve((size_t)24 * HDc * KNP * 2);
  float*    reltB  = (float*)   carve((size_t)24 * 6272 * 8 * 4);
  float*    relhB  = (float*)   carve((size_t)24 * 6272 * 7 * 4);
  float*    relwB  = (float*)   carve((size_t)24 * 6272 * 7 * 4);
  _Float16* ctx    = (_Float16*)carve((size_t)MPAD * DIM * 2);

  // 1) f32 -> f16 converts (pad rows of xh zeroed)
  {
    long nin = (long)MROWS * DIM, nout = (long)MPAD * DIM;
    f32_to_f16_pad<<<(int)((nout + 255) / 256), 256, 0, stream>>>(x, xh, nin, nout);
  }
  {
    long nw = (long)3 * DIM * DIM;
    f32_to_f16_pad<<<(int)((nw + 255) / 256), 256, 0, stream>>>(qkv_w, qkvwh, nw, nw);
  }
  {
    long nw = (long)DIM * DIM;
    f32_to_f16_pad<<<(int)((nw + 255) / 256), 256, 0, stream>>>(proj_w, projwh, nw, nw);
  }
  // zero ctx pad rows (never written by attention)
  {
    long np = (long)(MPAD - MROWS) * DIM;
    zero_f16<<<(int)((np + 255) / 256), 256, 0, stream>>>(
        ctx + (size_t)MROWS * DIM, np);
  }

  // 2) QKV GEMM: (12546 x 768) @ (2304 x 768)^T + qkv_b
  {
    dim3 g(MPAD / 32, (3 * DIM) / 64);
    wmma_gemm_bias<<<g, 32, 0, stream>>>(xh, qkvwh, qkv_b, qkv32,
                                         MROWS, DIM, 3 * DIM);
  }

  // 3) pool + LayerNorm
  {
    dim3 gq(MQP, 24);
    pool_ln_kernel<<<gq, 64, 0, stream>>>(qkv32, pq_w, nq_g, nq_b, qhB,
                                          0, 1, QTt, QHh, QWw, MQP, 0);
    dim3 gk(KNP, 24);
    pool_ln_kernel<<<gk, 64, 0, stream>>>(qkv32, pk_w, nk_g, nk_b, khB,
                                          1, 4, QTt, 7, 7, KNP, 0);
    dim3 gv(KNP, 24);
    pool_ln_kernel<<<gv, 64, 0, stream>>>(qkv32, pv_w, nv_g, nv_b, vthB,
                                          2, 4, QTt, 7, 7, KNP, 1);
  }

  // 4) rel-pos projections
  {
    dim3 gr(QTt * QHh * QWw, 24);
    relpos_kernel<<<gr, 32, 0, stream>>>(qhB, rph, rpw, rpt,
                                         reltB, relhB, relwB);
  }

  // 5) fused attention (QK + bias + softmax + AV), ctx f16
  {
    dim3 ga(MQP / 16, 24);
    attn_fused<<<ga, 32, 0, stream>>>(qhB, khB, vthB,
                                      reltB, relhB, relwB, ctx);
  }

  // 6) output projection: (12546 x 768) @ (768 x 768)^T + proj_b
  {
    dim3 g2(MPAD / 32, DIM / 64);
    wmma_gemm_bias<<<g2, 32, 0, stream>>>(ctx, projwh, proj_b, out,
                                          MROWS, DIM, DIM);
  }
}
